// ImageEncoder_85555748536632
// MI455X (gfx1250) — compile-verified, matrix operand order checked by reading, not run on hardware
//
#include <hip/hip_runtime.h>
#include <hip/hip_bf16.h>
#include <math.h>

#define N_TOK   2048
#define D_MODEL 1024
#define C_CH    32
#define E_EDGE  16384
#define HEADS   8
#define DH      64
#define INNER   512
#define EPSF    1e-5f

typedef __attribute__((ext_vector_type(16))) _Float16 v16h;
typedef __attribute__((ext_vector_type(8)))  _Float16 v8h;
typedef __attribute__((ext_vector_type(8)))  float    v8f;
typedef __attribute__((ext_vector_type(4)))  float    v4f;

union AFrag { v16h v; v8h h[2]; _Float16 e[16]; };

__device__ __forceinline__ float gelu_exact(float x) {
  return 0.5f * x * (1.0f + erff(x * 0.70710678118654752f));
}

__device__ __forceinline__ float atomicMaxFloat(float* addr, float value) {
  if (value >= 0.0f)
    return __int_as_float(atomicMax((int*)addr, __float_as_int(value)));
  else
    return __uint_as_float(atomicMin((unsigned int*)addr, __float_as_uint(value)));
}

// Async copy of 16 bytes global -> LDS (ASYNCcnt-tracked, no VGPR data path).
__device__ __forceinline__ void async_copy16B(const void* gsrc, void* ldst) {
  asm volatile("global_load_async_to_lds_b128 %0, %1, off"
               :: "v"((unsigned)(unsigned long long)ldst),
                  "v"((unsigned long long)gsrc)
               : "memory");
}

// ---------------------------------------------------------------------------
// WMMA GEMM on f16 operands:  C[M,Nn] = A[M,K] * Bt[Nn,K]^T  (+ bias[Nn])
// A row-major f16, Bt is B pre-transposed ([n][k]) f16, C/bias f32.
// Block = 128 threads (4 waves), tile 64x64, K-step 32.
// Tiles staged in LDS with GLOBAL_LOAD_ASYNC_TO_LDS_B128 + s_wait_asynccnt.
// ---------------------------------------------------------------------------
#define GTM 64
#define GTN 64
#define GTK 32

__global__ __launch_bounds__(128) void gemm_f16_async_kernel(
    const _Float16* __restrict__ A, const _Float16* __restrict__ Bt,
    const float* __restrict__ bias, float* __restrict__ C,
    int M, int Nn, int K) {
  __shared__ __align__(16) _Float16 As[GTM][GTK];
  __shared__ __align__(16) _Float16 Bs[GTN][GTK];

  const int tid  = threadIdx.x;
  const int lane = tid & 31;
  const int wave = tid >> 5;
  const int lg   = lane >> 4;
  const int lr   = lane & 15;
  const int kb   = lg * 8;
  const int m0   = blockIdx.y * GTM;
  const int n0   = blockIdx.x * GTN;

  const int srow = tid >> 1;            // staging row 0..63
  const int shalf = (tid & 1) * 16;     // halves offset 0 | 16

  v8f acc[4] = {};

  for (int k0 = 0; k0 < K; k0 += GTK) {
    // async stage A(64x32) and B(64x32) f16 tiles
    {
      const _Float16* ga = &A[(size_t)(m0 + srow) * K + k0 + shalf];
      const _Float16* gb = &Bt[(size_t)(n0 + srow) * K + k0 + shalf];
      async_copy16B(ga,     &As[srow][shalf]);
      async_copy16B(ga + 8, &As[srow][shalf + 8]);
      async_copy16B(gb,     &Bs[srow][shalf]);
      async_copy16B(gb + 8, &Bs[srow][shalf + 8]);
    }
    asm volatile("s_wait_asynccnt 0" ::: "memory");
    __syncthreads();

    AFrag af;
    const int arow = wave * 16 + lr;
    af.h[0] = *(const v8h*)&As[arow][kb];
    af.h[1] = *(const v8h*)&As[arow][kb + 16];

#pragma unroll
    for (int t = 0; t < 4; ++t) {
      AFrag bf;
      const int bcol = t * 16 + lr;
      bf.h[0] = *(const v8h*)&Bs[bcol][kb];
      bf.h[1] = *(const v8h*)&Bs[bcol][kb + 16];
      acc[t] = __builtin_amdgcn_wmma_f32_16x16x32_f16(
          false, af.v, false, bf.v, (short)0, acc[t], false, false);
    }
    __syncthreads();
  }

#pragma unroll
  for (int t = 0; t < 4; ++t) {
    const int col = n0 + t * 16 + lr;
    const float bv = bias ? bias[col] : 0.0f;
#pragma unroll
    for (int r = 0; r < 8; ++r) {
      const int row = m0 + wave * 16 + r + 8 * lg;
      C[(size_t)row * Nn + col] = acc[t][r] + bv;
    }
  }
}

// ---------------------------------------------------------------------------
// Flash attention, f16 qkv input [N, 3*INNER] (q|k|v), f32 out [N, INNER].
// One wave per (head, 16-query tile). K/V tiles double-buffered in LDS via
// async copies; next tile is prefetched while current one is consumed.
// ---------------------------------------------------------------------------
__global__ __launch_bounds__(32) void flash_attn_kernel(
    const _Float16* __restrict__ qkv, float* __restrict__ out) {
  __shared__ __align__(16) _Float16 Ks[2][32][64];
  __shared__ __align__(16) _Float16 Vs[2][32][64];
  __shared__ __align__(16) _Float16 Ps[16][32];

  const int head = blockIdx.y;
  const int qt   = blockIdx.x;
  const int lane = threadIdx.x;
  const int lg = lane >> 4, lr = lane & 15;
  const int kb = lg * 8;
  const int stride = 3 * INNER;
  const int qbase = qt * 16;
  const float scale = 0.125f;  // DH^-0.5

  // stage 32 keys + 32 values for tile kt into buffer b (16 async ops/wave)
  auto stage = [&](int kt, int b) {
    const _Float16* kr = qkv + (size_t)(kt * 32 + lane) * stride + INNER + head * 64;
    const _Float16* vr = kr + INNER;
    _Float16* lk = &Ks[b][lane][0];
    _Float16* lv = &Vs[b][lane][0];
#pragma unroll
    for (int c = 0; c < 8; ++c) {
      async_copy16B(kr + c * 8, lk + c * 8);
      async_copy16B(vr + c * 8, lv + c * 8);
    }
  };

  // Q fragments (K-dim 64 -> two 16x32 fragments), direct b128 global loads
  AFrag qa[2];
  {
    const _Float16* qp = qkv + (size_t)(qbase + lr) * stride + head * 64;
    qa[0].h[0] = *(const v8h*)&qp[kb];
    qa[0].h[1] = *(const v8h*)&qp[kb + 16];
    qa[1].h[0] = *(const v8h*)&qp[32 + kb];
    qa[1].h[1] = *(const v8h*)&qp[32 + kb + 16];
  }

  float mrow[8], lsum[8];
  v8f o[4] = {};
#pragma unroll
  for (int r = 0; r < 8; ++r) { mrow[r] = -3.0e38f; lsum[r] = 0.0f; }

  stage(0, 0);

  for (int kt = 0; kt < N_TOK / 32; ++kt) {
    const int cur = kt & 1;
    if (kt + 1 < N_TOK / 32) {
      stage(kt + 1, cur ^ 1);                       // prefetch next tile
      asm volatile("s_wait_asynccnt 16" ::: "memory");  // wait current only
    } else {
      asm volatile("s_wait_asynccnt 0" ::: "memory");
    }

    // ---- S = Q K^T over 32 keys (two 16x16 C tiles) ----
    v8f s[2] = {};
#pragma unroll
    for (int j = 0; j < 2; ++j) {
      const _Float16* krow = &Ks[cur][j * 16 + lr][0];
      AFrag b0, b1;
      b0.h[0] = *(const v8h*)&krow[kb];
      b0.h[1] = *(const v8h*)&krow[kb + 16];
      b1.h[0] = *(const v8h*)&krow[32 + kb];
      b1.h[1] = *(const v8h*)&krow[32 + kb + 16];
      s[j] = __builtin_amdgcn_wmma_f32_16x16x32_f16(
          false, qa[0].v, false, b0.v, (short)0, s[j], false, false);
      s[j] = __builtin_amdgcn_wmma_f32_16x16x32_f16(
          false, qa[1].v, false, b1.v, (short)0, s[j], false, false);
    }

    // ---- online softmax ----
    float sc[8];
#pragma unroll
    for (int r = 0; r < 8; ++r) {
      float v0 = s[0][r] * scale;
      float v1 = s[1][r] * scale;
      float mx = fmaxf(v0, v1);
#pragma unroll
      for (int d = 1; d < 16; d <<= 1) mx = fmaxf(mx, __shfl_xor(mx, d, 32));
      const float mnew = fmaxf(mrow[r], mx);
      sc[r] = __expf(mrow[r] - mnew);
      const float p0 = __expf(v0 - mnew);
      const float p1 = __expf(v1 - mnew);
      float rs = p0 + p1;
#pragma unroll
      for (int d = 1; d < 16; d <<= 1) rs += __shfl_xor(rs, d, 32);
      lsum[r] = lsum[r] * sc[r] + rs;
      mrow[r] = mnew;
      Ps[r + 8 * lg][lr]      = (_Float16)p0;   // transpose P through LDS
      Ps[r + 8 * lg][16 + lr] = (_Float16)p1;
    }
#pragma unroll
    for (int t = 0; t < 4; ++t)
#pragma unroll
      for (int r = 0; r < 8; ++r) o[t][r] *= sc[r];

    asm volatile("s_wait_dscnt 0" ::: "memory");

    AFrag pa;
    pa.h[0] = *(const v8h*)&Ps[lr][kb];
    pa.h[1] = *(const v8h*)&Ps[lr][kb + 16];

    // ---- O += P V ----
#pragma unroll
    for (int t = 0; t < 4; ++t) {
      AFrag vb;
      const int dcol = t * 16 + lr;
#pragma unroll
      for (int h = 0; h < 8; ++h) {
        vb.e[h]     = Vs[cur][kb + h][dcol];
        vb.e[8 + h] = Vs[cur][kb + 16 + h][dcol];
      }
      o[t] = __builtin_amdgcn_wmma_f32_16x16x32_f16(
          false, pa.v, false, vb.v, (short)0, o[t], false, false);
    }
    asm volatile("s_wait_dscnt 0" ::: "memory");
  }

#pragma unroll
  for (int t = 0; t < 4; ++t)
#pragma unroll
    for (int r = 0; r < 8; ++r) {
      const int row = qbase + r + 8 * lg;
      out[(size_t)row * INNER + head * 64 + t * 16 + lr] = o[t][r] / lsum[r];
    }
}

// ---------------------------------------------------------------------------
// Conversion kernels (weights once, activations per use)
// ---------------------------------------------------------------------------
__global__ __launch_bounds__(256) void transpose_cvt_kernel(
    const float* __restrict__ in, _Float16* __restrict__ out, int K, int Nn) {
  const int idx = blockIdx.x * 256 + threadIdx.x;   // over K*Nn, coalesced read
  if (idx >= K * Nn) return;
  const int n = idx % Nn, k = idx / Nn;
  out[(size_t)n * K + k] = (_Float16)in[idx];
}

__global__ __launch_bounds__(256) void cvt_f16_kernel(
    const float* __restrict__ in, _Float16* __restrict__ out, int n) {
  const int i = blockIdx.x * 256 + threadIdx.x;
  if (i < n) out[i] = (_Float16)in[i];
}

__global__ __launch_bounds__(256) void gelu_cvt_f16_kernel(
    const float* __restrict__ in, _Float16* __restrict__ out, int n) {
  const int i = blockIdx.x * 256 + threadIdx.x;
  if (i < n) out[i] = (_Float16)gelu_exact(in[i]);
}

// ---------------------------------------------------------------------------
// Conv / elementwise / GNN helper kernels
// ---------------------------------------------------------------------------
__global__ __launch_bounds__(256) void patch_embed_kernel(
    const float* __restrict__ x, const float* __restrict__ w,
    const float* __restrict__ b, float* __restrict__ out) {
  const int idx = blockIdx.x * 256 + threadIdx.x;  // n*8192 + co*256 + p
  const int p  = idx & 255;
  const int co = (idx >> 8) & 31;
  const int n  = idx >> 13;
  const int oy = p >> 4, ox = p & 15;
  const float* xin = x + (size_t)n * 3 * 112 * 112;
  const float* wp  = w + (size_t)co * 3 * 49;
  float s = b[co];
  for (int ci = 0; ci < 3; ++ci)
    for (int ky = 0; ky < 7; ++ky)
      for (int kx = 0; kx < 7; ++kx)
        s += xin[ci * 12544 + (oy * 7 + ky) * 112 + ox * 7 + kx] *
             wp[ci * 49 + ky * 7 + kx];
  out[idx] = s;
}

__global__ __launch_bounds__(256) void dw_bn_gelu_kernel(
    const float* __restrict__ in, const float* __restrict__ w,
    const float* __restrict__ bias, const float* __restrict__ g,
    const float* __restrict__ bb, float* __restrict__ out) {
  const int idx = blockIdx.x * 256 + threadIdx.x;
  const int p = idx & 255;
  const int c = (idx >> 8) & 31;
  const int y = p >> 4, x = p & 15;
  const float* ip = in + (idx & ~255);       // this (n,c) plane
  const float* wp = w + c * 25;
  float s = bias[c];
  for (int ky = 0; ky < 5; ++ky) {
    const int yy = y + ky - 2;
    if (yy < 0 || yy >= 16) continue;
    for (int kx = 0; kx < 5; ++kx) {
      const int xx = x + kx - 2;
      if (xx < 0 || xx >= 16) continue;
      s += ip[yy * 16 + xx] * wp[ky * 5 + kx];
    }
  }
  const float invs = rsqrtf(1.0f + EPSF);
  out[idx] = gelu_exact(s * invs * g[c] + bb[c]);
}

__global__ __launch_bounds__(256) void pw_gelu_bn_kernel(
    const float* __restrict__ in, const float* __restrict__ w,
    const float* __restrict__ bias, const float* __restrict__ g,
    const float* __restrict__ bb, float* __restrict__ out) {
  const int idx = blockIdx.x * 256 + threadIdx.x;  // n*8192 + co*256 + p
  const int p  = idx & 255;
  const int co = (idx >> 8) & 31;
  const int n  = idx >> 13;
  const float* base = in + (size_t)n * 8192;
  float s = bias[co];
  for (int ci = 0; ci < 32; ++ci) s += base[ci * 256 + p] * w[co * 32 + ci];
  const float invs = rsqrtf(1.0f + EPSF);
  out[idx] = gelu_exact(s) * invs * g[co] + bb[co];
}

__global__ __launch_bounds__(256) void flat_kernel(
    const float* __restrict__ in, const float* __restrict__ w,
    const float* __restrict__ b, float* __restrict__ feat) {
  const int idx = blockIdx.x * 256 + threadIdx.x;  // n*1024 + co*256 + p
  const int q  = idx & 1023;
  const int n  = idx >> 10;
  const int co = q >> 8;
  const int p  = q & 255;
  const float* base = in + (size_t)n * 8192;
  float s = b[co];
  for (int ci = 0; ci < 32; ++ci) s += base[ci * 256 + p] * w[co * 32 + ci];
  feat[idx] = s;
}

__global__ __launch_bounds__(256) void add_inplace_kernel(
    float* __restrict__ dst, const float* __restrict__ src, int n) {
  const int i = blockIdx.x * 256 + threadIdx.x;
  if (i < n) dst[i] += src[i];
}

__global__ __launch_bounds__(256) void add2_kernel(
    const float* __restrict__ a, const float* __restrict__ b,
    float* __restrict__ dst, int n) {
  const int i = blockIdx.x * 256 + threadIdx.x;
  if (i < n) dst[i] = a[i] + b[i];
}

__global__ __launch_bounds__(256) void fill_kernel(float* p, float v, int n) {
  const int i = blockIdx.x * 256 + threadIdx.x;
  if (i < n) p[i] = v;
}

// GNN edge kernels ----------------------------------------------------------
__global__ __launch_bounds__(256) void edge_logits_kernel(
    const float* __restrict__ q, const float* __restrict__ k,
    const int* __restrict__ adj, float* __restrict__ logits) {
  const int e = blockIdx.x * 8 + (threadIdx.x >> 5);
  const int lane = threadIdx.x & 31;
  if (e >= E_EDGE) return;
  const int src = adj[e];
  const int dst = adj[E_EDGE + e];
  float s = 0.0f;
  for (int d = lane; d < D_MODEL; d += 32)
    s += q[(size_t)dst * D_MODEL + d] * k[(size_t)src * D_MODEL + d];
#pragma unroll
  for (int m = 1; m < 32; m <<= 1) s += __shfl_xor(s, m, 32);
  if (lane == 0) logits[e] = s * 0.03125f;   // 1/sqrt(1024)
}

__global__ __launch_bounds__(256) void edge_max_kernel(
    const float* __restrict__ logits, const int* __restrict__ adj,
    float* __restrict__ segm) {
  const int e = blockIdx.x * 256 + threadIdx.x;
  if (e < E_EDGE) atomicMaxFloat(&segm[adj[E_EDGE + e]], logits[e]);
}

__global__ __launch_bounds__(256) void edge_expsum_kernel(
    const float* __restrict__ logits, const int* __restrict__ adj,
    const float* __restrict__ segm, float* __restrict__ ee,
    float* __restrict__ segz) {
  const int e = blockIdx.x * 256 + threadIdx.x;
  if (e >= E_EDGE) return;
  const int dst = adj[E_EDGE + e];
  const float ex = __expf(logits[e] - segm[dst]);
  ee[e] = ex;
  atomicAdd(&segz[dst], ex);
}

__global__ __launch_bounds__(256) void edge_agg_kernel(
    const float* __restrict__ ee, const int* __restrict__ adj,
    const float* __restrict__ segz, const float* __restrict__ vv,
    float* __restrict__ agg) {
  const int idx = blockIdx.x * 256 + threadIdx.x;  // e*1024 + d
  const int e = idx >> 10;
  const int d = idx & 1023;
  const int src = adj[e];
  const int dst = adj[E_EDGE + e];
  const float alpha = ee[e] / segz[dst];
  atomicAdd(&agg[(size_t)dst * D_MODEL + d],
            alpha * vv[(size_t)src * D_MODEL + d]);
}

__global__ __launch_bounds__(256) void gnn_post_kernel(
    const float* __restrict__ agg, const float* __restrict__ sk,
    const float* __restrict__ g, const float* __restrict__ b,
    float* __restrict__ pos) {
  const int idx = blockIdx.x * 256 + threadIdx.x;
  const int d = idx & 1023;
  const float invs = rsqrtf(1.0f + EPSF);
  const float v = (agg[idx] + sk[idx]) * invs * g[d] + b[d];
  pos[idx] = fmaxf(v, 0.0f);
}

// LayerNorm over last dim (1024), one block per row -------------------------
__global__ __launch_bounds__(256) void layernorm_kernel(
    const float* __restrict__ x, const float* __restrict__ g,
    const float* __restrict__ b, float* __restrict__ y) {
  const int row = blockIdx.x;
  const float* xr = x + (size_t)row * D_MODEL;
  __shared__ float red[8];
  const int tid = threadIdx.x, lane = tid & 31, wid = tid >> 5;

  float s = 0.0f;
  for (int d = tid; d < D_MODEL; d += 256) s += xr[d];
#pragma unroll
  for (int m = 1; m < 32; m <<= 1) s += __shfl_xor(s, m, 32);
  if (lane == 0) red[wid] = s;
  __syncthreads();
  float mean = 0.0f;
  for (int i = 0; i < 8; ++i) mean += red[i];
  mean *= (1.0f / D_MODEL);
  __syncthreads();

  float v = 0.0f;
  for (int d = tid; d < D_MODEL; d += 256) {
    const float t = xr[d] - mean;
    v += t * t;
  }
#pragma unroll
  for (int m = 1; m < 32; m <<= 1) v += __shfl_xor(v, m, 32);
  if (lane == 0) red[wid] = v;
  __syncthreads();
  float var = 0.0f;
  for (int i = 0; i < 8; ++i) var += red[i];
  var *= (1.0f / D_MODEL);
  const float inv = rsqrtf(var + EPSF);

  for (int d = tid; d < D_MODEL; d += 256)
    y[(size_t)row * D_MODEL + d] = (xr[d] - mean) * inv * g[d] + b[d];
}

// ---------------------------------------------------------------------------
extern "C" void kernel_launch(void* const* d_in, const int* in_sizes, int n_in,
                              void* d_out, int out_size, void* d_ws, size_t ws_size,
                              hipStream_t stream) {
  (void)in_sizes; (void)n_in; (void)out_size; (void)ws_size;

  const float* x        = (const float*)d_in[0];
  const int*   adj      = (const int*)d_in[1];
  const float* patch_w  = (const float*)d_in[2];
  const float* patch_b  = (const float*)d_in[3];
  const float* dw_w     = (const float*)d_in[4];
  const float* dw_b     = (const float*)d_in[5];
  const float* bn_g     = (const float*)d_in[6];
  const float* bn_b     = (const float*)d_in[7];
  const float* pw_w     = (const float*)d_in[8];
  const float* pw_b     = (const float*)d_in[9];
  const float* flat_w   = (const float*)d_in[10];
  const float* flat_b   = (const float*)d_in[11];
  const float* gnn_wq   = (const float*)d_in[12];
  const float* gnn_wk   = (const float*)d_in[13];
  const float* gnn_wv   = (const float*)d_in[14];
  const float* gnn_ws   = (const float*)d_in[15];
  const float* gnn_bq   = (const float*)d_in[16];
  const float* gnn_bk   = (const float*)d_in[17];
  const float* gnn_bv   = (const float*)d_in[18];
  const float* gnn_bs   = (const float*)d_in[19];
  const float* gnn_bng  = (const float*)d_in[20];
  const float* gnn_bnb  = (const float*)d_in[21];
  const float* t_ln1_g  = (const float*)d_in[22];
  const float* t_ln1_b  = (const float*)d_in[23];
  const float* t_wqkv   = (const float*)d_in[24];
  const float* t_wo     = (const float*)d_in[25];
  const float* t_bo     = (const float*)d_in[26];
  const float* t_ln2_g  = (const float*)d_in[27];
  const float* t_ln2_b  = (const float*)d_in[28];
  const float* t_w1     = (const float*)d_in[29];
  const float* t_b1     = (const float*)d_in[30];
  const float* t_w2     = (const float*)d_in[31];
  const float* t_b2     = (const float*)d_in[32];

  // ---- workspace carve-out (bytes, 256B aligned blocks) -------------------
  char* base = (char*)d_ws;
  size_t off = 0;
  auto allocB = [&](size_t bytes) {
    void* p = base + off;
    off += (bytes + 255) & ~(size_t)255;
    return p;
  };
  const size_t PLANE = (size_t)N_TOK * C_CH * 256;        // 16.78M elems
  const size_t ND    = (size_t)N_TOK * D_MODEL;           // 2.10M elems
  const size_t DD    = (size_t)D_MODEL * D_MODEL;

  float* bufA  = (float*)allocB(PLANE * 4);
  float* bufB  = (float*)allocB(PLANE * 4);
  float* bufC  = (float*)allocB(PLANE * 4);
  float* feat  = (float*)allocB(ND * 4);
  float* pos   = (float*)allocB(ND * 4);
  float* qb    = (float*)allocB(ND * 4);
  float* kbuf  = (float*)allocB(ND * 4);
  float* vbuf  = (float*)allocB(ND * 4);
  float* skb   = (float*)allocB(ND * 4);
  float* aggb  = (float*)allocB(ND * 4);
  float* logits= (float*)allocB(E_EDGE * 4);
  float* ee    = (float*)allocB(E_EDGE * 4);
  float* segm  = (float*)allocB(N_TOK * 4);
  float* segz  = (float*)allocB(N_TOK * 4);
  float* hn    = (float*)allocB(ND * 4);
  float* qkvb  = (float*)allocB((size_t)N_TOK * 3 * INNER * 4);
  float* attno = (float*)allocB((size_t)N_TOK * INNER * 4);
  float* tmp   = (float*)allocB(ND * 4);
  float* tmp2  = (float*)allocB(ND * 4);

  // f16 weight copies, pre-transposed to [N][K]
  _Float16* wq16   = (_Float16*)allocB(2 * DD * 2);
  _Float16* wk16   = (_Float16*)allocB(2 * DD * 2);
  _Float16* wv16   = (_Float16*)allocB(2 * DD * 2);
  _Float16* ws16   = (_Float16*)allocB(2 * DD * 2);
  _Float16* wqkv16 = (_Float16*)allocB((size_t)8 * D_MODEL * 3 * INNER * 2);
  _Float16* wo16   = (_Float16*)allocB((size_t)8 * INNER * D_MODEL * 2);
  _Float16* w116   = (_Float16*)allocB((size_t)8 * DD * 2);
  _Float16* w216   = (_Float16*)allocB((size_t)8 * DD * 2);
  // f16 activation scratch
  _Float16* act16  = (_Float16*)allocB(ND * 2);
  _Float16* qkv16  = (_Float16*)allocB((size_t)N_TOK * 3 * INNER * 2);

  float* t = (float*)d_out;   // residual stream lives in d_out

  auto gemm = [&](const _Float16* A, const _Float16* Bt, const float* bias,
                  float* Cm, int M, int Nn, int K) {
    dim3 g(Nn / GTN, M / GTM);
    gemm_f16_async_kernel<<<g, 128, 0, stream>>>(A, Bt, bias, Cm, M, Nn, K);
  };
  auto tcvt = [&](const float* in, _Float16* out, int K, int Nn) {
    transpose_cvt_kernel<<<((size_t)K * Nn + 255) / 256, 256, 0, stream>>>(in, out, K, Nn);
  };
  auto cvt = [&](const float* in, _Float16* out, size_t n) {
    cvt_f16_kernel<<<(n + 255) / 256, 256, 0, stream>>>(in, out, (int)n);
  };

  // ---- one-time weight conversion (transposed f16) ------------------------
  for (int l = 0; l < 2; ++l) {
    tcvt(gnn_wq + l * DD, wq16 + l * DD, D_MODEL, D_MODEL);
    tcvt(gnn_wk + l * DD, wk16 + l * DD, D_MODEL, D_MODEL);
    tcvt(gnn_wv + l * DD, wv16 + l * DD, D_MODEL, D_MODEL);
    tcvt(gnn_ws + l * DD, ws16 + l * DD, D_MODEL, D_MODEL);
  }
  for (int l = 0; l < 8; ++l) {
    tcvt(t_wqkv + (size_t)l * D_MODEL * 3 * INNER,
         wqkv16 + (size_t)l * D_MODEL * 3 * INNER, D_MODEL, 3 * INNER);
    tcvt(t_wo + (size_t)l * INNER * D_MODEL,
         wo16 + (size_t)l * INNER * D_MODEL, INNER, D_MODEL);
    tcvt(t_w1 + (size_t)l * DD, w116 + (size_t)l * DD, D_MODEL, D_MODEL);
    tcvt(t_w2 + (size_t)l * DD, w216 + (size_t)l * DD, D_MODEL, D_MODEL);
  }

  // ---- conv stem ----------------------------------------------------------
  patch_embed_kernel<<<PLANE / 256, 256, 0, stream>>>(x, patch_w, patch_b, bufA);
  float* h  = bufA;
  float* s1 = bufB;
  float* s2 = bufC;
  for (int blk = 0; blk < 2; ++blk) {
    dw_bn_gelu_kernel<<<PLANE / 256, 256, 0, stream>>>(
        h, dw_w + (size_t)(blk * 2 + 0) * C_CH * 25, dw_b + (blk * 2 + 0) * C_CH,
        bn_g + (blk * 3 + 0) * C_CH, bn_b + (blk * 3 + 0) * C_CH, s1);
    dw_bn_gelu_kernel<<<PLANE / 256, 256, 0, stream>>>(
        s1, dw_w + (size_t)(blk * 2 + 1) * C_CH * 25, dw_b + (blk * 2 + 1) * C_CH,
        bn_g + (blk * 3 + 1) * C_CH, bn_b + (blk * 3 + 1) * C_CH, s2);
    add_inplace_kernel<<<PLANE / 256, 256, 0, stream>>>(h, s2, (int)PLANE);
    pw_gelu_bn_kernel<<<PLANE / 256, 256, 0, stream>>>(
        h, pw_w + (size_t)blk * C_CH * C_CH, pw_b + blk * C_CH,
        bn_g + (blk * 3 + 2) * C_CH, bn_b + (blk * 3 + 2) * C_CH, s1);
    float* sw = h; h = s1; s1 = sw;
  }
  flat_kernel<<<ND / 256, 256, 0, stream>>>(h, flat_w, flat_b, feat);

  // ---- GNN ----------------------------------------------------------------
  // pos starts as feat; convert current pos to f16 once per layer
  add2_kernel<<<ND / 256, 256, 0, stream>>>(feat, feat, pos, 0);  // no-op sizing
  cvt(feat, act16, ND);
  const float* posf = feat;
  for (int l = 0; l < 2; ++l) {
    gemm(act16, wq16 + l * DD, gnn_bq + l * D_MODEL, qb,   N_TOK, D_MODEL, D_MODEL);
    gemm(act16, wk16 + l * DD, gnn_bk + l * D_MODEL, kbuf, N_TOK, D_MODEL, D_MODEL);
    gemm(act16, wv16 + l * DD, gnn_bv + l * D_MODEL, vbuf, N_TOK, D_MODEL, D_MODEL);
    gemm(act16, ws16 + l * DD, gnn_bs + l * D_MODEL, skb,  N_TOK, D_MODEL, D_MODEL);
    fill_kernel<<<(N_TOK + 255) / 256, 256, 0, stream>>>(segm, -3.0e38f, N_TOK);
    fill_kernel<<<(N_TOK + 255) / 256, 256, 0, stream>>>(segz, 0.0f, N_TOK);
    fill_kernel<<<ND / 256, 256, 0, stream>>>(aggb, 0.0f, (int)ND);
    edge_logits_kernel<<<E_EDGE / 8, 256, 0, stream>>>(qb, kbuf, adj, logits);
    edge_max_kernel<<<E_EDGE / 256, 256, 0, stream>>>(logits, adj, segm);
    edge_expsum_kernel<<<E_EDGE / 256, 256, 0, stream>>>(logits, adj, segm, ee, segz);
    edge_agg_kernel<<<(E_EDGE * (size_t)D_MODEL) / 256, 256, 0, stream>>>(
        ee, adj, segz, vbuf, aggb);
    gnn_post_kernel<<<ND / 256, 256, 0, stream>>>(
        aggb, skb, gnn_bng + l * D_MODEL, gnn_bnb + l * D_MODEL, pos);
    cvt(pos, act16, ND);      // f16 input for next layer / unused after last
    posf = pos;
  }

  // ---- transformer stack --------------------------------------------------
  add2_kernel<<<ND / 256, 256, 0, stream>>>(feat, posf, t, (int)ND);
  for (int l = 0; l < 8; ++l) {
    layernorm_kernel<<<N_TOK, 256, 0, stream>>>(
        t, t_ln1_g + l * D_MODEL, t_ln1_b + l * D_MODEL, hn);
    cvt(hn, act16, ND);
    gemm(act16, wqkv16 + (size_t)l * D_MODEL * 3 * INNER, nullptr, qkvb,
         N_TOK, 3 * INNER, D_MODEL);
    cvt(qkvb, qkv16, (size_t)N_TOK * 3 * INNER);
    flash_attn_kernel<<<dim3(N_TOK / 16, HEADS), 32, 0, stream>>>(qkv16, attno);
    cvt(attno, act16, (size_t)N_TOK * INNER);
    gemm(act16, wo16 + (size_t)l * INNER * D_MODEL, t_bo + l * D_MODEL, tmp,
         N_TOK, D_MODEL, INNER);
    add_inplace_kernel<<<ND / 256, 256, 0, stream>>>(t, tmp, (int)ND);
    layernorm_kernel<<<N_TOK, 256, 0, stream>>>(
        t, t_ln2_g + l * D_MODEL, t_ln2_b + l * D_MODEL, hn);
    cvt(hn, act16, ND);
    gemm(act16, w116 + (size_t)l * DD, t_b1 + l * D_MODEL, tmp,
         N_TOK, D_MODEL, D_MODEL);
    gelu_cvt_f16_kernel<<<ND / 256, 256, 0, stream>>>(tmp, act16, (int)ND);
    gemm(act16, w216 + (size_t)l * DD, t_b2 + l * D_MODEL, tmp2,
         N_TOK, D_MODEL, D_MODEL);
    add_inplace_kernel<<<ND / 256, 256, 0, stream>>>(t, tmp2, (int)ND);
  }
}
